// SelfAttention_7696581394823
// MI455X (gfx1250) — matrix operand order repaired and checked by reading, not compiled, not run on hardware
//
#include <hip/hip_runtime.h>
#include <hip/hip_bf16.h>

// Problem constants (from reference): x [B, C, 64, 64]
#define Bn 4
#define Cc 256
#define Nn 4096   // 64*64

typedef __attribute__((ext_vector_type(16))) __bf16    v16bf;
typedef __attribute__((ext_vector_type(8)))  float     v8f;
typedef __attribute__((ext_vector_type(4)))  unsigned  v4u;
typedef __attribute__((ext_vector_type(8)))  int       v8i;
typedef __attribute__((ext_vector_type(4)))  int       v4i;

// ---------------------------------------------------------------- helpers

__device__ __forceinline__ unsigned short f2bf(float f) {
    unsigned u = __builtin_bit_cast(unsigned, f);
    unsigned r = u + 0x7fffu + ((u >> 16) & 1u);   // round-to-nearest-even
    return (unsigned short)(r >> 16);
}

// A-fragment (16x32 bf16): lane holds row (lane&15); two 16B chunks at
// K = lhalf*8 and K = 16 + lhalf*8.  `prow` points at [row][k0].
__device__ __forceinline__ v16bf load_frag_a(const unsigned short* prow, int lhalf) {
    v16bf f;
    ((uint4*)&f)[0] = *(const uint4*)(prow + lhalf * 8);
    ((uint4*)&f)[1] = *(const uint4*)(prow + 16 + lhalf * 8);
    return f;
}

// B-fragment (32x16 bf16): lane holds column (lane&15); two 16B chunks at
// K = lhalf*8 and K = 16 + lhalf*8.  Source layout is [col][k] (k contiguous).
__device__ __forceinline__ v16bf load_frag_b(const unsigned short* pcol, int lhalf) {
    v16bf f;
    ((uint4*)&f)[0] = *(const uint4*)(pcol + lhalf * 8);
    ((uint4*)&f)[1] = *(const uint4*)(pcol + 16 + lhalf * 8);
    return f;
}

__device__ __forceinline__ v8f wmma_bf16(v16bf a, v16bf b, v8f c) {
    return __builtin_amdgcn_wmma_f32_16x16x32_bf16(false, a, false, b,
                                                   (short)0, c, false, false);
}

// LDS byte address = low 32 bits of the generic (flat) pointer (ISA 10.2).
__device__ __forceinline__ unsigned lds_addr32(const void* p) {
    return (unsigned)(uintptr_t)p;
}

// Tensor Data Mover: 2-D tile load Global -> LDS.  data_size = 8 bytes.
// D# per ISA ch.8: group0 = {count, lds_addr, global_addr, type=2};
// group1 = {data_size, tensor_dim0/1, tile_dim0/1, tensor_dim0_stride}.
// Dims/strides are in 8-byte units.  Tracked with TENSORcnt.
__device__ __forceinline__ void tdm_load_2d(unsigned lds_byte, const void* gptr,
                                            unsigned tile_d0, unsigned tile_d1,
                                            unsigned tensor_d0, unsigned tensor_d1,
                                            unsigned long long stride0_units) {
    unsigned long long ga = (unsigned long long)(uintptr_t)gptr;
    v4u g0 = { 1u,                                   // count = 1 (valid)
               lds_byte,                             // lds_addr
               (unsigned)ga,                         // global_addr[31:0]
               (unsigned)((ga >> 32) & 0x01FFFFFFu)  // global_addr[56:32]
                   | 0x80000000u };                  // type = 2 ("image")
    v8i g1 = { (int)(3u << 16),                                            // data_size = 8B
               (int)((tensor_d0 & 0xFFFFu) << 16),                         // tensor_dim0[15:0]
               (int)((tensor_d0 >> 16) | ((tensor_d1 & 0xFFFFu) << 16)),   // td0[31:16], td1[15:0]
               (int)((tensor_d1 >> 16) | (tile_d0 << 16)),                 // td1[31:16], tile_dim0
               (int)tile_d1,                                               // tile_dim1 (tile_dim2=0)
               (int)(unsigned)(stride0_units & 0xFFFFFFFFull),             // stride0[31:0]
               (int)(unsigned)((stride0_units >> 32) & 0xFFFFull),         // stride0[47:32]
               0 };
    v4i z4 = {0, 0, 0, 0};
#if defined(__clang_major__) && __clang_major__ >= 23
    v8i z8 = {0, 0, 0, 0, 0, 0, 0, 0};
    __builtin_amdgcn_tensor_load_to_lds(g0, g1, z4, z4, z8, 0);
#else
    __builtin_amdgcn_tensor_load_to_lds(g0, g1, z4, z4, 0);
#endif
}

// ---------------------------------------------------------------- K0a: weight f32 -> bf16

__global__ void k_wcvt(const float* __restrict__ a, const float* __restrict__ b,
                       const float* __restrict__ c,
                       unsigned short* __restrict__ oa, unsigned short* __restrict__ ob,
                       unsigned short* __restrict__ oc) {
    int i = blockIdx.x * 256 + threadIdx.x;
    oa[i] = f2bf(a[i]);
    ob[i] = f2bf(b[i]);
    oc[i] = f2bf(c[i]);
}

// ---------------------------------------------------------------- K0b: x [B][C][N] f32 -> xT [B][N][C] bf16

__global__ void k_transpose(const float* __restrict__ x, unsigned short* __restrict__ xT) {
    __shared__ float tile[32][33];
    int n0 = blockIdx.x * 32, c0 = blockIdx.y * 32, b = blockIdx.z;
    const float* xb = x + (size_t)b * Cc * Nn;
#pragma unroll
    for (int i = 0; i < 4; i++) {
        int c = threadIdx.y + i * 8;
        tile[c][threadIdx.x] = xb[(size_t)(c0 + c) * Nn + n0 + threadIdx.x];
    }
    __syncthreads();
    unsigned short* o = xT + (size_t)b * Nn * Cc;
#pragma unroll
    for (int i = 0; i < 4; i++) {
        int n = threadIdx.y + i * 8;
        o[(size_t)(n0 + n) * Cc + c0 + threadIdx.x] = f2bf(tile[threadIdx.x][n]);
    }
}

// ---------------------------------------------------------------- K1: QKV projections (bf16 WMMA)
// Each wave computes a 16x64 tile; K-loop over C=256 is software-pipelined
// (fragments for step kk+1 load while step kk's WMMAs issue).

__global__ void __launch_bounds__(256) k_qkv(
        const unsigned short* __restrict__ xT,
        const unsigned short* __restrict__ Wq, const unsigned short* __restrict__ Wk,
        const unsigned short* __restrict__ Wv,
        const float* __restrict__ bq, const float* __restrict__ bk,
        const float* __restrict__ bv,
        unsigned short* __restrict__ q, unsigned short* __restrict__ k,
        unsigned short* __restrict__ v) {
    int wid  = threadIdx.x >> 5;
    int lane = threadIdx.x & 31;
    int lrow = lane & 15, lhalf = lane >> 4;

    int gid  = blockIdx.x * 8 + wid;   // 0..12287
    int type = gid >> 12;              // 0=q, 1=k, 2=v
    int t    = gid & 4095;
    int b    = t >> 10;
    int rem  = t & 1023;

    v8f acc[4] = {v8f{}, v8f{}, v8f{}, v8f{}};

    const unsigned short* Abase;
    const unsigned short* Bbase;
    size_t bstride;
    if (type < 2) {
        // rows = n, cols = c_out.  A = xT[n][c], B = W[c_out][c].
        int n0 = (rem >> 2) * 16;
        int c0 = (rem & 3) * 64;
        const unsigned short* W = (type == 0) ? Wq : Wk;
        Abase   = xT + ((size_t)b * Nn + n0 + lrow) * Cc;
        Bbase   = W + (size_t)(c0 + lrow) * Cc;
        bstride = (size_t)16 * Cc;
    } else {
        // rows = c_out, cols = n.  A = Wv[c_out][c], B = xT[n][c].
        int co0 = (rem >> 6) * 16;
        int n0  = (rem & 63) * 64;
        Abase   = Wv + (size_t)(co0 + lrow) * Cc;
        Bbase   = xT + ((size_t)b * Nn + n0 + lrow) * Cc;
        bstride = (size_t)16 * Cc;
    }

    // software-pipelined K loop
    v16bf a = load_frag_a(Abase, lhalf);
    v16bf bb[4];
#pragma unroll
    for (int j = 0; j < 4; j++) bb[j] = load_frag_b(Bbase + j * bstride, lhalf);
#pragma unroll
    for (int kk = 0; kk < 8; kk++) {
        v16bf an;
        v16bf bn[4];
        if (kk < 7) {
            an = load_frag_a(Abase + (kk + 1) * 32, lhalf);
#pragma unroll
            for (int j = 0; j < 4; j++)
                bn[j] = load_frag_b(Bbase + j * bstride + (kk + 1) * 32, lhalf);
        }
#pragma unroll
        for (int j = 0; j < 4; j++) acc[j] = wmma_bf16(a, bb[j], acc[j]);
        if (kk < 7) {
            a = an;
#pragma unroll
            for (int j = 0; j < 4; j++) bb[j] = bn[j];
        }
    }

    if (type < 2) {
        int n0 = (rem >> 2) * 16;
        int c0 = (rem & 3) * 64;
        const float*    bias = (type == 0) ? bq : bk;
        unsigned short* dst  = (type == 0) ? q : k;
        float scale = (type == 0) ? 0.0625f : 1.0f;   // 1/sqrt(256) folded into q
#pragma unroll
        for (int j = 0; j < 4; j++) {
            int   cc = c0 + j * 16 + lrow;
            float bj = bias[cc];
#pragma unroll
            for (int r = 0; r < 8; r++) {
                int n = n0 + lhalf * 8 + r;
                dst[((size_t)b * Nn + n) * Cc + cc] = f2bf((acc[j][r] + bj) * scale);
            }
        }
    } else {
        int co0 = (rem >> 6) * 16;
        int n0  = (rem & 63) * 64;
#pragma unroll
        for (int j = 0; j < 4; j++) {
            int nn = n0 + j * 16 + lrow;
#pragma unroll
            for (int r = 0; r < 8; r++) {
                int c = co0 + lhalf * 8 + r;
                v[((size_t)b * Cc + c) * Nn + nn] = f2bf(acc[j][r] + bv[c]);
            }
        }
    }
}

// ---------------------------------------------------------------- K2: fused flash attention
// Workgroup-cooperative: wave0/wave1 drive the Tensor Data Mover to stage the
// 32-key K tile (contiguous 16KB) and V tile (256 x 64B, strided) into LDS,
// double-buffered, tracked with TENSORcnt.  All 8 waves consume B-fragments
// from LDS.  Each wave owns 16 query rows with online softmax in registers.

__global__ void __launch_bounds__(256, 1) k_attn(
        const unsigned short* __restrict__ q, const unsigned short* __restrict__ k,
        const unsigned short* __restrict__ v, float* __restrict__ out) {
    __shared__ unsigned short kbuf[2][32 * 256];   // [m_local][c]   16KB each
    __shared__ unsigned short vbuf[2][256 * 32];   // [c][m_local]   16KB each
    __shared__ unsigned short plds[8][16 * 32];    // wave-private P tiles

    int wid  = threadIdx.x >> 5;
    int lane = threadIdx.x & 31;
    int lrow = lane & 15, lhalf = lane >> 4;

    int b  = blockIdx.x >> 5;                           // 32 blocks per batch
    int n0 = (((blockIdx.x & 31) << 3) + wid) << 4;     // 16-row query block per wave

    const unsigned short* qb = q + ((size_t)b * Nn + n0) * Cc;
    const unsigned short* kb = k + (size_t)b * Nn * Cc;
    const unsigned short* vb = v + (size_t)b * Cc * Nn;

    // prologue: stage tile m0=0 into buffer 0 via TDM
    if (wid == 0)
        tdm_load_2d(lds_addr32(&kbuf[0][0]), kb, 2048, 1, 2048, 1, 2048);
    else if (wid == 1)
        tdm_load_2d(lds_addr32(&vbuf[0][0]), vb, 8, 256, 8, 256, 1024);

    // Q block resident in registers: 8 A-fragments (16x256 bf16)
    v16bf qf[8];
#pragma unroll
    for (int kk = 0; kk < 8; kk++)
        qf[kk] = load_frag_a(qb + (size_t)lrow * Cc + kk * 32, lhalf);

    v8f o[16];
#pragma unroll
    for (int j = 0; j < 16; j++) o[j] = v8f{};
    float mrun[8], lrun[8];
#pragma unroll
    for (int r = 0; r < 8; r++) { mrun[r] = -1e30f; lrun[r] = 0.0f; }

    unsigned short* myp = plds[wid];

    for (int m0 = 0; m0 < Nn; m0 += 32) {
        int buf = (m0 >> 5) & 1;
        // issue next tile's DMA first, then wait for the current tile
        if (wid == 0) {
            if (m0 + 32 < Nn) {
                tdm_load_2d(lds_addr32(&kbuf[buf ^ 1][0]), kb + (size_t)(m0 + 32) * Cc,
                            2048, 1, 2048, 1, 2048);
                __builtin_amdgcn_s_wait_tensorcnt(1);
            } else {
                __builtin_amdgcn_s_wait_tensorcnt(0);
            }
        } else if (wid == 1) {
            if (m0 + 32 < Nn) {
                tdm_load_2d(lds_addr32(&vbuf[buf ^ 1][0]), vb + m0 + 32,
                            8, 256, 8, 256, 1024);
                __builtin_amdgcn_s_wait_tensorcnt(1);
            } else {
                __builtin_amdgcn_s_wait_tensorcnt(0);
            }
        }
        __syncthreads();   // K/V tile visible to all waves

        const unsigned short* kt = kbuf[buf];
        const unsigned short* vt = vbuf[buf];

        // ---- scores: two 16x16 tiles over K = 256, B-frags from LDS
        v8f s0 = v8f{}, s1 = v8f{};
#pragma unroll
        for (int kk = 0; kk < 8; kk++) {
            v16bf b0 = load_frag_b(kt + (size_t)lrow * Cc + kk * 32, lhalf);
            v16bf b1 = load_frag_b(kt + (size_t)(16 + lrow) * Cc + kk * 32, lhalf);
            s0 = wmma_bf16(qf[kk], b0, s0);
            s1 = wmma_bf16(qf[kk], b1, s1);
        }
        // ---- online softmax; D-frag: lane = col m, reg r = row n (lhalf*8+r)
        float alpha[8];
#pragma unroll
        for (int r = 0; r < 8; r++) {
            float mx = fmaxf(s0[r], s1[r]);
            mx = fmaxf(mx, __shfl_xor(mx, 1, 32));
            mx = fmaxf(mx, __shfl_xor(mx, 2, 32));
            mx = fmaxf(mx, __shfl_xor(mx, 4, 32));
            mx = fmaxf(mx, __shfl_xor(mx, 8, 32));
            float mnew = fmaxf(mrun[r], mx);
            float a    = __expf(mrun[r] - mnew);
            float p0   = __expf(s0[r] - mnew);
            float p1   = __expf(s1[r] - mnew);
            float ls   = p0 + p1;
            ls += __shfl_xor(ls, 1, 32);
            ls += __shfl_xor(ls, 2, 32);
            ls += __shfl_xor(ls, 4, 32);
            ls += __shfl_xor(ls, 8, 32);
            lrun[r]  = lrun[r] * a + ls;
            mrun[r]  = mnew;
            alpha[r] = a;
            int row = lhalf * 8 + r;
            myp[row * 32 + lrow]      = f2bf(p0);
            myp[row * 32 + 16 + lrow] = f2bf(p1);
        }
        // ---- rescale running output (rows align with softmax stats)
#pragma unroll
        for (int j = 0; j < 16; j++)
#pragma unroll
            for (int r = 0; r < 8; r++)
                o[j][r] *= alpha[r];
        // ---- relayout P (D-frag -> A-frag) through wave-private LDS
        v16bf pf;
        ((uint4*)&pf)[0] = *(const uint4*)&myp[lrow * 32 + lhalf * 8];
        ((uint4*)&pf)[1] = *(const uint4*)&myp[lrow * 32 + 16 + lhalf * 8];
        // ---- O[n, c] += P[n, m] * V^T[m, c], B-frags from LDS V tile
#pragma unroll
        for (int j = 0; j < 16; j++) {
            v16bf vf = load_frag_b(vt + (size_t)(j * 16 + lrow) * 32, lhalf);
            o[j] = wmma_bf16(pf, vf, o[j]);
        }
        __syncthreads();   // all waves done with this buffer before DMA reuses it
    }

    // ---- epilogue: divide by row sums, store out[b][c][n] (n-contiguous per lane)
    float inv[8];
#pragma unroll
    for (int r = 0; r < 8; r++) inv[r] = 1.0f / lrun[r];
    float* ob = out + (size_t)b * Cc * Nn;
#pragma unroll
    for (int j = 0; j < 16; j++) {
        int    c  = j * 16 + lrow;
        float4 t0 = {o[j][0] * inv[0], o[j][1] * inv[1], o[j][2] * inv[2], o[j][3] * inv[3]};
        float4 t1 = {o[j][4] * inv[4], o[j][5] * inv[5], o[j][6] * inv[6], o[j][7] * inv[7]};
        float* p = ob + (size_t)c * Nn + n0 + lhalf * 8;
        *(float4*)p       = t0;
        *(float4*)(p + 4) = t1;
    }
}

// ---------------------------------------------------------------- launch

extern "C" void kernel_launch(void* const* d_in, const int* in_sizes, int n_in,
                              void* d_out, int out_size, void* d_ws, size_t ws_size,
                              hipStream_t stream) {
    const float* x  = (const float*)d_in[0];
    const float* Wq = (const float*)d_in[1];
    const float* bq = (const float*)d_in[2];
    const float* Wk = (const float*)d_in[3];
    const float* bk = (const float*)d_in[4];
    const float* Wv = (const float*)d_in[5];
    const float* bv = (const float*)d_in[6];
    float* out = (float*)d_out;

    char*  ws  = (char*)d_ws;
    size_t off = 0;
    auto alloc = [&](size_t bytes) -> char* {
        char* p = ws + off;
        off = (off + bytes + 255) & ~(size_t)255;
        return p;
    };
    unsigned short* xT  = (unsigned short*)alloc((size_t)Bn * Nn * Cc * 2);
    unsigned short* qws = (unsigned short*)alloc((size_t)Bn * Nn * Cc * 2);
    unsigned short* kws = (unsigned short*)alloc((size_t)Bn * Nn * Cc * 2);
    unsigned short* vws = (unsigned short*)alloc((size_t)Bn * Cc * Nn * 2);
    unsigned short* Wqb = (unsigned short*)alloc((size_t)Cc * Cc * 2);
    unsigned short* Wkb = (unsigned short*)alloc((size_t)Cc * Cc * 2);
    unsigned short* Wvb = (unsigned short*)alloc((size_t)Cc * Cc * 2);

    k_wcvt<<<(Cc * Cc) / 256, 256, 0, stream>>>(Wq, Wk, Wv, Wqb, Wkb, Wvb);
    k_transpose<<<dim3(Nn / 32, Cc / 32, Bn), dim3(32, 8), 0, stream>>>(x, xT);
    k_qkv<<<1536, 256, 0, stream>>>(xT, Wqb, Wkb, Wvb, bq, bk, bv, qws, kws, vws);
    k_attn<<<128, 256, 0, stream>>>(qws, kws, vws, out);
}
